// SAGELayer1_41240275976352
// MI455X (gfx1250) — matrix operand order repaired and checked by reading, not compiled, not run on hardware
//
#include <hip/hip_runtime.h>
#include <hip/hip_bf16.h>

typedef __attribute__((ext_vector_type(16))) __bf16 v16bf;
typedef __attribute__((ext_vector_type(8)))  float  v8f;

#define IN_CH 128
#define HID_CH 128
#define WMAT_ELEMS 16384          // 128x128 halfs per packed matrix
#define WPACK_ELEMS (4 * WMAT_ELEMS)  // WlHi|WlLo|WrHi|WrLo

// ---------------------------------------------------------------------------
// Zero a float region of d_ws (harness poisons ws with 0xAA).
// ---------------------------------------------------------------------------
__global__ void zero_f32(float* __restrict__ p, size_t n) {
    size_t i = (size_t)blockIdx.x * blockDim.x + threadIdx.x;
    size_t stride = (size_t)gridDim.x * blockDim.x;
    for (; i < n; i += stride) p[i] = 0.0f;
}

// ---------------------------------------------------------------------------
// Pre-swizzle a 128x128 row-major f32 weight matrix into the CDNA5 WMMA
// B-operand layout (bf16, hi + lo error-compensation terms).
//
// Packed layout: flat index t = (((nt*4 + c)*32 + lane)*16 + e)
//   nt = N-tile (0..7), c = K-chunk (0..3), lane = wave lane, e = element.
// B 32x16 (KxN), contiguous k order used here: lanes 0-15 hold K=0..15 for N=lane,
//                               lanes 16-31 hold K=16..31 for N=lane-16.
// ---------------------------------------------------------------------------
__global__ void pack_wmma_b(const float* __restrict__ W,
                            __bf16* __restrict__ hi,
                            __bf16* __restrict__ lo) {
    int t = blockIdx.x * blockDim.x + threadIdx.x;
    if (t >= WMAT_ELEMS) return;
    int e    = t & 15;
    int lane = (t >> 4) & 31;
    int c    = (t >> 9) & 3;
    int nt   = t >> 11;
    int k = c * 32 + ((lane >= 16) ? 16 : 0) + e;
    int n = nt * 16 + (lane & 15);
    float w = W[k * HID_CH + n];
    __bf16 h = (__bf16)w;
    hi[t] = h;
    lo[t] = (__bf16)(w - (float)h);
}

// ---------------------------------------------------------------------------
// Edge scatter: agg[dst] += x[src] (f32 atomics), cnt[dst] += 1.
// 32 lanes per edge, float4 per lane -> 4 global_atomic_add_f32 each.
// Gather of x[src] is a fully coalesced 512B read per edge.
// ---------------------------------------------------------------------------
__global__ void scatter_mean_accum(const float* __restrict__ x,
                                   const int* __restrict__ src,
                                   const int* __restrict__ dst,
                                   float* __restrict__ agg,
                                   float* __restrict__ cnt,
                                   int n_edges) {
    size_t gid = (size_t)blockIdx.x * blockDim.x + threadIdx.x;
    size_t total = (size_t)n_edges * 32;
    if (gid >= total) return;
    int e = (int)(gid >> 5);
    int g = (int)(gid & 31);           // channel group (4 floats each)
    int s = src[e];
    int d = dst[e];
    float4 v = ((const float4*)(x + (size_t)s * IN_CH))[g];
    float* a = agg + (size_t)d * IN_CH + g * 4;
    atomicAdd(a + 0, v.x);
    atomicAdd(a + 1, v.y);
    atomicAdd(a + 2, v.z);
    atomicAdd(a + 3, v.w);
    if (g == 0) atomicAdd(cnt + d, 1.0f);
}

// ---------------------------------------------------------------------------
// Fused GEMM: out = relu( (agg/cnt) @ W_l + x @ W_r + b )
// One wave32 per 16-row M-tile; 8 waves/block. K=128 -> 4 chunks of 32.
// bf16 hi/lo split: A@B ~= Ahi@Bhi + Ahi@Blo + Alo@Bhi (near-fp32 accuracy).
// Packed weights (128KB) staged once per block into LDS; inner-loop B
// operands come from ds_load_b128 instead of redundant global reads (8x cut).
// 6 WMMAs per (chunk, N-tile) -> 192 v_wmma_f32_16x16x32_bf16 per wave.
// ---------------------------------------------------------------------------
__global__ void __launch_bounds__(256) sage_gemm(
        const float* __restrict__ x,
        const float* __restrict__ agg,
        const float* __restrict__ cnt,
        const float* __restrict__ bias,
        const __bf16* __restrict__ Wpack,   // WlHi|WlLo|WrHi|WrLo
        float* __restrict__ out, int n_nodes) {
    alignas(32) __shared__ __bf16 sW[WPACK_ELEMS];   // 128 KB

    // Cooperative fill: 8192 x uint4 (16B), 32 per thread, coalesced.
    {
        const uint4* gsrc = (const uint4*)Wpack;
        uint4* ldst = (uint4*)sW;
        const int tid = threadIdx.x;
#pragma unroll
        for (int i = 0; i < 32; ++i) {
            ldst[tid + i * 256] = gsrc[tid + i * 256];
        }
    }
    __syncthreads();   // all waves participate before any wave exits

    const int wave = threadIdx.x >> 5;
    const int lane = threadIdx.x & 31;
    const int tileM = blockIdx.x * 8 + wave;
    if (tileM * 16 >= n_nodes) return;           // wave-uniform exit (post-barrier)

    const __bf16* sWlHi = sW;
    const __bf16* sWlLo = sW + WMAT_ELEMS;
    const __bf16* sWrHi = sW + 2 * WMAT_ELEMS;
    const __bf16* sWrLo = sW + 3 * WMAT_ELEMS;

    // A-operand row for this lane (ISA 16-bit A layout: lanes 0-15 and 16-31
    // both cover M=0..15 but different K halves).
    const int row   = tileM * 16 + (lane & 15);
    const int khalf = (lane >= 16) ? 16 : 0;

    const float cv  = cnt[row];
    const float inv = 1.0f / fmaxf(cv, 1.0f);
    const float* xr = x   + (size_t)row * IN_CH;
    const float* ar = agg + (size_t)row * IN_CH;
    __builtin_prefetch(xr, 0, 3);   // global_prefetch_b8
    __builtin_prefetch(ar, 0, 3);

    v8f acc[8];
#pragma unroll
    for (int i = 0; i < 8; ++i) acc[i] = (v8f)(0.0f);

#pragma unroll
    for (int c = 0; c < 4; ++c) {
        const int kb = c * 32 + khalf;
        // A layout per lane: elems 0..15 -> K=kb+e
        float fx[16], fa[16];
        {
            float4 t0 = *(const float4*)(xr + kb);
            float4 t1 = *(const float4*)(xr + kb + 4);
            float4 t2 = *(const float4*)(xr + kb + 8);
            float4 t3 = *(const float4*)(xr + kb + 12);
            fx[0]=t0.x; fx[1]=t0.y; fx[2]=t0.z; fx[3]=t0.w;
            fx[4]=t1.x; fx[5]=t1.y; fx[6]=t1.z; fx[7]=t1.w;
            fx[8]=t2.x; fx[9]=t2.y; fx[10]=t2.z; fx[11]=t2.w;
            fx[12]=t3.x; fx[13]=t3.y; fx[14]=t3.z; fx[15]=t3.w;
            float4 a0 = *(const float4*)(ar + kb);
            float4 a1 = *(const float4*)(ar + kb + 4);
            float4 a2 = *(const float4*)(ar + kb + 8);
            float4 a3 = *(const float4*)(ar + kb + 12);
            fa[0]=a0.x; fa[1]=a0.y; fa[2]=a0.z; fa[3]=a0.w;
            fa[4]=a1.x; fa[5]=a1.y; fa[6]=a1.z; fa[7]=a1.w;
            fa[8]=a2.x; fa[9]=a2.y; fa[10]=a2.z; fa[11]=a2.w;
            fa[12]=a3.x; fa[13]=a3.y; fa[14]=a3.z; fa[15]=a3.w;
        }
        v16bf xhi, xlo, mhi, mlo;
#pragma unroll
        for (int i = 0; i < 16; ++i) {
            float fv = fx[i];
            __bf16 h = (__bf16)fv;
            xhi[i] = h;
            xlo[i] = (__bf16)(fv - (float)h);
            float mv = fa[i] * inv;
            __bf16 mh = (__bf16)mv;
            mhi[i] = mh;
            mlo[i] = (__bf16)(mv - (float)mh);
        }
#pragma unroll
        for (int nt = 0; nt < 8; ++nt) {
            const size_t boff = (((size_t)nt * 4 + c) * 32 + lane) * 16;
            v16bf blh = *(const v16bf*)(sWlHi + boff);   // ds_load_b128 x2
            v16bf bll = *(const v16bf*)(sWlLo + boff);
            v16bf brh = *(const v16bf*)(sWrHi + boff);
            v16bf brl = *(const v16bf*)(sWrLo + boff);
            v8f a = acc[nt];
            a = __builtin_amdgcn_wmma_f32_16x16x32_bf16(false, mhi, false, blh, (short)0, a, false, false);
            a = __builtin_amdgcn_wmma_f32_16x16x32_bf16(false, mhi, false, bll, (short)0, a, false, false);
            a = __builtin_amdgcn_wmma_f32_16x16x32_bf16(false, mlo, false, blh, (short)0, a, false, false);
            a = __builtin_amdgcn_wmma_f32_16x16x32_bf16(false, xhi, false, brh, (short)0, a, false, false);
            a = __builtin_amdgcn_wmma_f32_16x16x32_bf16(false, xhi, false, brl, (short)0, a, false, false);
            a = __builtin_amdgcn_wmma_f32_16x16x32_bf16(false, xlo, false, brh, (short)0, a, false, false);
            acc[nt] = a;
        }
    }

    // Epilogue: C/D layout -> lane gives N (l&15), element r gives M
    // (lanes 16-31 carry M=8..15). Add bias, relu, store f32.
    const int ncol_base = lane & 15;
    const int mbase = tileM * 16 + ((lane >= 16) ? 8 : 0);
#pragma unroll
    for (int nt = 0; nt < 8; ++nt) {
        const int ncol = nt * 16 + ncol_base;
        const float bv = bias[ncol];
#pragma unroll
        for (int r = 0; r < 8; ++r) {
            float v = acc[nt][r] + bv;
            out[(size_t)(mbase + r) * HID_CH + ncol] = fmaxf(v, 0.0f);
        }
    }
}

// ---------------------------------------------------------------------------
// Launch: zero ws -> pack weights -> scatter atomics -> fused WMMA GEMM.
// ws layout: agg [N*128 f32] | cnt [N f32] | Wpack [4*16384 bf16]
// ---------------------------------------------------------------------------
extern "C" void kernel_launch(void* const* d_in, const int* in_sizes, int n_in,
                              void* d_out, int out_size, void* d_ws, size_t ws_size,
                              hipStream_t stream) {
    const float* x  = (const float*)d_in[0];
    const int*   ei = (const int*)d_in[1];
    const float* Wl = (const float*)d_in[2];
    const float* Wr = (const float*)d_in[3];
    const float* b  = (const float*)d_in[4];

    const int n_nodes = in_sizes[0] / IN_CH;   // 100000
    const int n_edges = in_sizes[1] / 2;       // 1600000
    const int* src = ei;
    const int* dst = ei + n_edges;

    float* agg = (float*)d_ws;
    float* cnt = agg + (size_t)n_nodes * IN_CH;
    __bf16* wpack = (__bf16*)(cnt + n_nodes);
    __bf16* WlHi = wpack;
    __bf16* WlLo = wpack + WMAT_ELEMS;
    __bf16* WrHi = wpack + 2 * WMAT_ELEMS;
    __bf16* WrLo = wpack + 3 * WMAT_ELEMS;

    // 1) zero agg + cnt
    size_t nz = (size_t)n_nodes * IN_CH + (size_t)n_nodes;
    zero_f32<<<2048, 256, 0, stream>>>(agg, nz);

    // 2) pack weights into WMMA B-operand bf16 hi/lo layout
    pack_wmma_b<<<(WMAT_ELEMS + 255) / 256, 256, 0, stream>>>(Wl, WlHi, WlLo);
    pack_wmma_b<<<(WMAT_ELEMS + 255) / 256, 256, 0, stream>>>(Wr, WrHi, WrLo);

    // 3) edge scatter-add (mean numerator + degree counts)
    size_t total = (size_t)n_edges * 32;
    int sblocks = (int)((total + 255) / 256);
    scatter_mean_accum<<<sblocks, 256, 0, stream>>>(x, src, dst, agg, cnt, n_edges);

    // 4) fused mean + dual-GEMM + bias + relu via bf16 WMMA (hi/lo split),
    //    weights fed from LDS.
    const int m_tiles = (n_nodes + 15) / 16;           // 6250
    const int gblocks = (m_tiles + 7) / 8;             // 8 waves/block
    sage_gemm<<<gblocks, 256, 0, stream>>>(x, agg, cnt, b, wpack,
                                           (float*)d_out, n_nodes);
}
